// MoE_77678778516066
// MI455X (gfx1250) — matrix operand order repaired and checked by reading, not compiled, not run on hardware
//
#include <hip/hip_runtime.h>
#include <hip/hip_bf16.h>
#include <stdint.h>

// ---- problem constants (match reference) ----
#define B_TOK  2048
#define D_IN   1024
#define O_OUT  1024
#define NE     8
#define HDIM   10240   // D*10

typedef unsigned int u32;
typedef __attribute__((ext_vector_type(4)))  u32            u32x4;
typedef __attribute__((ext_vector_type(4)))  unsigned short u16x4;
typedef __attribute__((ext_vector_type(16))) __bf16         v16bf;
typedef __attribute__((ext_vector_type(8)))  float          v8f;
typedef int v4i_vs __attribute__((vector_size(16)));   // matches builtin param

struct Frag32B { u32x4 lo, hi; };   // 32 bytes = one 16x16x32 bf16 A or B fragment

// ---- CDNA5 async global->LDS path (guarded; falls back to reg staging) ----
#if defined(__has_builtin)
#if __has_builtin(__builtin_amdgcn_global_load_async_to_lds_b128)
#define USE_ASYNC_LDS 1
#endif
#endif

#ifdef USE_ASYNC_LDS
#if __has_builtin(__builtin_amdgcn_s_wait_asynccnt)
#define WAIT_ASYNC() __builtin_amdgcn_s_wait_asynccnt(0)
#else
#define WAIT_ASYNC() asm volatile("s_wait_asynccnt 0" ::: "memory")
#endif
static __device__ __forceinline__ void cp_async16(u32* lds, const unsigned short* g) {
  __builtin_amdgcn_global_load_async_to_lds_b128(
      (__attribute__((address_space(1))) v4i_vs*)(g),
      (__attribute__((address_space(3))) v4i_vs*)(lds), 0, 0);
}
#endif

static __device__ __forceinline__ unsigned short f2bf(float f) {
  u32 u = __float_as_uint(f);
  u32 r = u + 0x7fffu + ((u >> 16) & 1u);   // round-to-nearest-even
  return (unsigned short)(r >> 16);
}

// ---------------- fp32 -> bf16 bulk convert (x4 vectorized) ----------------
__global__ void cvt_f32_bf16(const float* __restrict__ src,
                             unsigned short* __restrict__ dst, int n4) {
  int i = blockIdx.x * blockDim.x + threadIdx.x;
  int stride = gridDim.x * blockDim.x;
  for (; i < n4; i += stride) {
    float4 f = ((const float4*)src)[i];
    u16x4 o;
    o[0] = f2bf(f.x); o[1] = f2bf(f.y); o[2] = f2bf(f.z); o[3] = f2bf(f.w);
    ((u16x4*)dst)[i] = o;
  }
}

// ---------------- gate: logits -> softmax -> top2 -> scatter ----------------
__global__ void gate_topk(const float* __restrict__ x1,
                          const float* __restrict__ gw,
                          const float* __restrict__ gb,
                          float* __restrict__ gate_scaled,
                          int* __restrict__ rowlist,
                          int* __restrict__ counts) {
  const int lane = threadIdx.x & 31;
  const int b = blockIdx.x * (blockDim.x >> 5) + (threadIdx.x >> 5);
  if (b >= B_TOK) return;

  float p[NE];
#pragma unroll
  for (int e = 0; e < NE; ++e) p[e] = 0.f;

  for (int d = lane; d < D_IN; d += 32) {
    float x = x1[(size_t)b * D_IN + d];
#pragma unroll
    for (int e = 0; e < NE; ++e) p[e] += x * gw[(size_t)e * D_IN + d];
  }
#pragma unroll
  for (int e = 0; e < NE; ++e) {
#pragma unroll
    for (int off = 16; off > 0; off >>= 1) p[e] += __shfl_xor(p[e], off, 32);
    p[e] += gb[e];
  }
  if (lane == 0) {
    float m = p[0];
#pragma unroll
    for (int e = 1; e < NE; ++e) m = fmaxf(m, p[e]);
    float pr[NE], s = 0.f;
#pragma unroll
    for (int e = 0; e < NE; ++e) { pr[e] = __expf(p[e] - m); s += pr[e]; }
    float inv = 1.f / s;
#pragma unroll
    for (int e = 0; e < NE; ++e) pr[e] *= inv;
    int i1 = 0;
#pragma unroll
    for (int e = 1; e < NE; ++e) if (pr[e] > pr[i1]) i1 = e;
    int i2 = (i1 == 0) ? 1 : 0;
#pragma unroll
    for (int e = 0; e < NE; ++e) if (e != i1 && pr[e] > pr[i2]) i2 = e;
#pragma unroll
    for (int e = 0; e < NE; ++e)
      gate_scaled[b * NE + e] = (e == i1) ? pr[i1] : ((e == i2) ? pr[i2] : 0.f);
    int p1 = atomicAdd(&counts[i1], 1); rowlist[i1 * B_TOK + p1] = b;
    int p2 = atomicAdd(&counts[i2], 1); rowlist[i2 * B_TOK + p2] = b;
  }
}

// ---------------- out[b,o] = sum_e gate[b,e] * fc2_b[e,o] ----------------
__global__ void bias_init(const float* __restrict__ gs,
                          const float* __restrict__ fc2_b,
                          float* __restrict__ out) {
  int i = blockIdx.x * blockDim.x + threadIdx.x;
  int stride = gridDim.x * blockDim.x;
  for (; i < B_TOK * O_OUT; i += stride) {
    int b = i >> 10, o = i & 1023;
    float acc = 0.f;
#pragma unroll
    for (int e = 0; e < NE; ++e) acc += gs[b * NE + e] * fc2_b[e * O_OUT + o];
    out[i] = acc;
  }
}

// ======================= WMMA tile compute =======================
// LDS tile: 128 rows x 32 bf16 (16 dwords/row), K-pairs packed per dword, which
// matches the ISA 16-bit A(16x32)/B(32x16) fragment layouts.
static __device__ __forceinline__ void wmma_tile(const u32* __restrict__ sAbuf,
                                                 const u32* __restrict__ sBbuf,
                                                 int Mb, int Nb, int lh, int lm,
                                                 v8f c[4][2]) {
  Frag32B afr[4], bfr[2];
#pragma unroll
  for (int mi = 0; mi < 4; ++mi) {
    int m = Mb + mi * 16 + lm;
    afr[mi].lo = *(const u32x4*)&sAbuf[m * 16 + lh * 4];
    afr[mi].hi = *(const u32x4*)&sAbuf[m * 16 + 8 + lh * 4];
  }
#pragma unroll
  for (int ni = 0; ni < 2; ++ni) {
    int n = Nb + ni * 16 + lm;
    bfr[ni].lo = *(const u32x4*)&sBbuf[n * 16 + lh * 4];
    bfr[ni].hi = *(const u32x4*)&sBbuf[n * 16 + 8 + lh * 4];
  }
#pragma unroll
  for (int mi = 0; mi < 4; ++mi)
#pragma unroll
    for (int ni = 0; ni < 2; ++ni)
      c[mi][ni] = __builtin_amdgcn_wmma_f32_16x16x32_bf16(
          false, __builtin_bit_cast(v16bf, afr[mi]),
          false, __builtin_bit_cast(v16bf, bfr[ni]),
          (short)0, c[mi][ni], false, false);
}

// Double-buffered, x2-unrolled K pipeline with constant buffer indices.
// Requires KDIM % 64 == 0.  Each buffer is only overwritten after a barrier
// that follows (in every wave's program order) all reads of that buffer.
#ifdef USE_ASYNC_LDS
#define K_PIPELINE(KDIM)                                                       \
  STAGE(0, 0);                                                                 \
  _Pragma("nounroll")                                                          \
  for (int kk = 0; kk < (KDIM); kk += 64) {                                    \
    WAIT_ASYNC(); __syncthreads();                                             \
    STAGE(kk + 32, 1);                                                         \
    wmma_tile(sA[0], sB[0], Mb, Nb, lh, lm, c);                                \
    WAIT_ASYNC(); __syncthreads();                                             \
    if (kk + 64 < (KDIM)) STAGE(kk + 64, 0);                                   \
    wmma_tile(sA[1], sB[1], Mb, Nb, lh, lm, c);                                \
  }
#else
#define K_PIPELINE(KDIM)                                                       \
  LOADR(0); STORE_LDS(0);                                                      \
  _Pragma("nounroll")                                                          \
  for (int kk = 0; kk < (KDIM); kk += 64) {                                    \
    __syncthreads();                                                           \
    LOADR(kk + 32);                                                            \
    wmma_tile(sA[0], sB[0], Mb, Nb, lh, lm, c);                                \
    STORE_LDS(1);                                                              \
    __syncthreads();                                                           \
    if (kk + 64 < (KDIM)) LOADR(kk + 64);                                      \
    wmma_tile(sA[1], sB[1], Mb, Nb, lh, lm, c);                                \
    if (kk + 64 < (KDIM)) STORE_LDS(0);                                        \
  }
#endif

// ---------------- GEMM1: hiddenS = gate * relu(x2g @ W1^T + b1), bf16 ----------------
__global__ __launch_bounds__(256)
void moe_gemm1(const unsigned short* __restrict__ x2bf,
               const unsigned short* __restrict__ w1bf,
               const float* __restrict__ fc1_b,
               const float* __restrict__ gate_scaled,
               const int* __restrict__ rowlist,
               const int* __restrict__ counts,
               unsigned short* __restrict__ hiddenS) {
  const int e  = blockIdx.x >> 4;
  const int mt = blockIdx.x & 15;
  const int cnt = counts[e];
  if (mt * 128 >= cnt) return;
  const int nblk = blockIdx.y * 128;

  __shared__ __align__(16) u32 sA[2][128 * 16];
  __shared__ __align__(16) u32 sB[2][128 * 16];
  __shared__ int   sRow[128];
  __shared__ float sG[128];

  const int tid = threadIdx.x;
  if (tid < 128) {
    int gi = mt * 128 + tid;
    int r = (gi < cnt) ? rowlist[e * B_TOK + gi] : 0;
    sRow[tid] = r;
    sG[tid] = (gi < cnt) ? gate_scaled[r * NE + e] : 0.f;
  }
  __syncthreads();

  const int wv = tid >> 5, lane = tid & 31;
  const int wr = wv & 1, wc = wv >> 1;           // waves: 2 (M) x 4 (N)
  const int Mb = wr * 64, Nb = wc * 32;
  const int lh = lane >> 4, lm = lane & 15;

  v8f c[4][2] = {};

  const int ldrow = tid >> 1, ldhalf = tid & 1;
  const unsigned short* aSrc =
      x2bf + (size_t)sRow[ldrow] * D_IN + ldhalf * 16;
  const unsigned short* bSrc =
      w1bf + (size_t)e * HDIM * D_IN + (size_t)(nblk + ldrow) * D_IN + ldhalf * 16;
  const int dstDw = ldrow * 16 + ldhalf * 8;

#ifdef USE_ASYNC_LDS
#define STAGE(KOFF, BUF)                                                        \
  do {                                                                          \
    cp_async16(&sA[BUF][dstDw],     aSrc + (KOFF));                             \
    cp_async16(&sA[BUF][dstDw + 4], aSrc + (KOFF) + 8);                         \
    cp_async16(&sB[BUF][dstDw],     bSrc + (KOFF));                             \
    cp_async16(&sB[BUF][dstDw + 4], bSrc + (KOFF) + 8);                         \
  } while (0)
  K_PIPELINE(D_IN)
#undef STAGE
#else
  u32x4 ra0, ra1, rb0, rb1;
#define LOADR(KOFF)                                                             \
  do {                                                                          \
    ra0 = ((const u32x4*)(aSrc + (KOFF)))[0];                                   \
    ra1 = ((const u32x4*)(aSrc + (KOFF)))[1];                                   \
    rb0 = ((const u32x4*)(bSrc + (KOFF)))[0];                                   \
    rb1 = ((const u32x4*)(bSrc + (KOFF)))[1];                                   \
  } while (0)
#define STORE_LDS(BUF)                                                          \
  do {                                                                          \
    *(u32x4*)&sA[BUF][dstDw] = ra0; *(u32x4*)&sA[BUF][dstDw + 4] = ra1;         \
    *(u32x4*)&sB[BUF][dstDw] = rb0; *(u32x4*)&sB[BUF][dstDw + 4] = rb1;         \
  } while (0)
  K_PIPELINE(D_IN)
#undef LOADR
#undef STORE_LDS
#endif

#pragma unroll
  for (int mi = 0; mi < 4; ++mi) {
#pragma unroll
    for (int ni = 0; ni < 2; ++ni) {
      int hcol = nblk + Nb + ni * 16 + lm;
      float bias = fc1_b[e * HDIM + hcol];
#pragma unroll
      for (int r = 0; r < 8; ++r) {
        int mrow = Mb + mi * 16 + r + lh * 8;
        float v = c[mi][ni][r] + bias;
        v = fmaxf(v, 0.f) * sG[mrow];
        hiddenS[((size_t)(e * B_TOK + mt * 128 + mrow)) * HDIM + hcol] = f2bf(v);
      }
    }
  }
}

// ---------------- GEMM2: out[b,:] += hiddenS_row @ W2^T (atomic scatter) ----------------
__global__ __launch_bounds__(256)
void moe_gemm2(const unsigned short* __restrict__ hiddenS,
               const unsigned short* __restrict__ w2bf,
               const int* __restrict__ rowlist,
               const int* __restrict__ counts,
               float* __restrict__ out) {
  const int e  = blockIdx.x >> 4;
  const int mt = blockIdx.x & 15;
  const int cnt = counts[e];
  if (mt * 128 >= cnt) return;
  const int nblk = blockIdx.y * 128;

  __shared__ __align__(16) u32 sA[2][128 * 16];
  __shared__ __align__(16) u32 sB[2][128 * 16];
  __shared__ int sRow[128];

  const int tid = threadIdx.x;
  if (tid < 128) {
    int gi = mt * 128 + tid;
    sRow[tid] = (gi < cnt) ? rowlist[e * B_TOK + gi] : 0;
  }
  __syncthreads();

  const int wv = tid >> 5, lane = tid & 31;
  const int wr = wv & 1, wc = wv >> 1;
  const int Mb = wr * 64, Nb = wc * 32;
  const int lh = lane >> 4, lm = lane & 15;

  v8f c[4][2] = {};

  const int ldrow = tid >> 1, ldhalf = tid & 1;
  const unsigned short* aSrc =
      hiddenS + (size_t)(e * B_TOK + mt * 128 + ldrow) * HDIM + ldhalf * 16;
  const unsigned short* bSrc =
      w2bf + (size_t)e * O_OUT * HDIM + (size_t)(nblk + ldrow) * HDIM + ldhalf * 16;
  const int dstDw = ldrow * 16 + ldhalf * 8;

#ifdef USE_ASYNC_LDS
#define STAGE(KOFF, BUF)                                                        \
  do {                                                                          \
    cp_async16(&sA[BUF][dstDw],     aSrc + (KOFF));                             \
    cp_async16(&sA[BUF][dstDw + 4], aSrc + (KOFF) + 8);                         \
    cp_async16(&sB[BUF][dstDw],     bSrc + (KOFF));                             \
    cp_async16(&sB[BUF][dstDw + 4], bSrc + (KOFF) + 8);                         \
  } while (0)
  K_PIPELINE(HDIM)
#undef STAGE
#else
  u32x4 ra0, ra1, rb0, rb1;
#define LOADR(KOFF)                                                             \
  do {                                                                          \
    ra0 = ((const u32x4*)(aSrc + (KOFF)))[0];                                   \
    ra1 = ((const u32x4*)(aSrc + (KOFF)))[1];                                   \
    rb0 = ((const u32x4*)(bSrc + (KOFF)))[0];                                   \
    rb1 = ((const u32x4*)(bSrc + (KOFF)))[1];                                   \
  } while (0)
#define STORE_LDS(BUF)                                                          \
  do {                                                                          \
    *(u32x4*)&sA[BUF][dstDw] = ra0; *(u32x4*)&sA[BUF][dstDw + 4] = ra1;         \
    *(u32x4*)&sB[BUF][dstDw] = rb0; *(u32x4*)&sB[BUF][dstDw + 4] = rb1;         \
  } while (0)
  K_PIPELINE(HDIM)
#undef LOADR
#undef STORE_LDS
#endif

#pragma unroll
  for (int mi = 0; mi < 4; ++mi) {
#pragma unroll
    for (int ni = 0; ni < 2; ++ni) {
      int ocol = nblk + Nb + ni * 16 + lm;
#pragma unroll
      for (int r = 0; r < 8; ++r) {
        int mrow = Mb + mi * 16 + r + lh * 8;
        if (mt * 128 + mrow < cnt) {
          int b = sRow[mrow];
          atomicAdd(&out[(size_t)b * O_OUT + ocol], c[mi][ni][r]);
        }
      }
    }
  }
}

// ======================= host-side launch =======================
extern "C" void kernel_launch(void* const* d_in, const int* in_sizes, int n_in,
                              void* d_out, int out_size, void* d_ws, size_t ws_size,
                              hipStream_t stream) {
  const float* x1     = (const float*)d_in[0];
  const float* x2     = (const float*)d_in[1];
  const float* gate_w = (const float*)d_in[2];
  const float* gate_b = (const float*)d_in[3];
  const float* fc1_w  = (const float*)d_in[4];
  const float* fc1_b  = (const float*)d_in[5];
  const float* fc2_w  = (const float*)d_in[6];
  const float* fc2_b  = (const float*)d_in[7];
  float* out = (float*)d_out;

  char* ws = (char*)d_ws;
  const size_t off_counts = 0;                                  // 256 B
  const size_t off_gs     = 256;                                // B*NE*4
  const size_t off_rl     = off_gs + (size_t)B_TOK * NE * 4;    // NE*B*4
  const size_t off_x2     = off_rl + (size_t)NE * B_TOK * 4;
  const size_t off_w1     = off_x2 + (size_t)B_TOK * D_IN * 2;
  const size_t off_w2     = off_w1 + (size_t)NE * HDIM * D_IN * 2;
  const size_t off_hid    = off_w2 + (size_t)NE * O_OUT * HDIM * 2;

  int*   counts = (int*)(ws + off_counts);
  float* gs     = (float*)(ws + off_gs);
  int*   rl     = (int*)(ws + off_rl);
  unsigned short* x2bf = (unsigned short*)(ws + off_x2);
  unsigned short* w1bf = (unsigned short*)(ws + off_w1);
  unsigned short* w2bf = (unsigned short*)(ws + off_w2);
  unsigned short* hid  = (unsigned short*)(ws + off_hid);

  (void)hipMemsetAsync(counts, 0, 256, stream);

  // fp32 -> bf16 conversions
  cvt_f32_bf16<<<2048, 256, 0, stream>>>(x2, x2bf, (B_TOK * D_IN) / 4);
  cvt_f32_bf16<<<4096, 256, 0, stream>>>(fc1_w, w1bf, (NE * HDIM * D_IN) / 4);
  cvt_f32_bf16<<<4096, 256, 0, stream>>>(fc2_w, w2bf, (NE * O_OUT * HDIM) / 4);

  // gating + routing
  gate_topk<<<B_TOK / 8, 256, 0, stream>>>(x1, gate_w, gate_b, gs, rl, counts);

  // gated bias term initializes the output
  bias_init<<<2048, 256, 0, stream>>>(gs, fc2_b, out);

  // expert GEMMs (gathered, top-2 only)
  moe_gemm1<<<dim3(NE * 16, HDIM / 128), 256, 0, stream>>>(
      x2bf, w1bf, fc1_b, gs, rl, counts, hid);
  moe_gemm2<<<dim3(NE * 16, O_OUT / 128), 256, 0, stream>>>(
      hid, w2bf, rl, counts, out);
}
